// MaacCritic_24464133718607
// MI455X (gfx1250) — matrix operand order repaired and checked by reading, not compiled, not run on hardware
//
#include <hip/hip_runtime.h>
#include <hip/hip_bf16.h>
#include <stdint.h>

// ---------------- problem constants ----------------
#define A_N   8
#define B_N   1024
#define T_N   32
#define NTOK  (B_N * T_N)      // 32768 tokens per agent
#define OBS   64
#define ACT   16
#define IDIM  80               // OBS + ACT
#define H     128
#define HEADS 4
#define DD    32               // head dim

typedef __attribute__((ext_vector_type(16))) _Float16 v16h;
typedef __attribute__((ext_vector_type(8)))  float    v8f;

// ---------------- workspace layout (units: halves) ----------------
constexpr size_t N_ENCW = (size_t)8 * 3 * 8 * 512;   // 98304
constexpr size_t N_SW   = (size_t)8 * 2 * 8 * 512;   // 65536
constexpr size_t N_HW   = (size_t)4 * 8 * 512;       // 16384 (per head-proj weight)
constexpr size_t N_CW1  = (size_t)8 * 8 * 8 * 512;   // 262144
constexpr size_t N_CW2  = (size_t)8 * 4 * 512;       // 16384
constexpr size_t N_ACTB = (size_t)A_N * NTOK * H;    // 33554432 per activation buffer

constexpr size_t OFF_ENCW  = 0;
constexpr size_t OFF_SW    = OFF_ENCW + N_ENCW;
constexpr size_t OFF_KEYW  = OFF_SW   + N_SW;
constexpr size_t OFF_SELW  = OFF_KEYW + N_HW;
constexpr size_t OFF_VALW  = OFF_SELW + N_HW;
constexpr size_t OFF_CW1   = OFF_VALW + N_HW;
constexpr size_t OFF_CW2   = OFF_CW1  + N_CW1;
constexpr size_t OFF_SE    = OFF_CW2  + N_CW2;       // f16 [A][NTOK][H]
constexpr size_t OFF_KEYS  = OFF_SE   + N_ACTB;
constexpr size_t OFF_SEL   = OFF_KEYS + N_ACTB;
constexpr size_t OFF_VALS  = OFF_SEL  + N_ACTB;
constexpr size_t OFF_OTHER = OFF_VALS + N_ACTB;

#define ATTN_SCALE 0.17677669529663687f   // 1/sqrt(DD), folded into sel_W at prep

// ---------------- WMMA fragment helpers (wave32, 16x16x32 f16) ----------------
// A-operand: lane holds row m = lane&15; element e -> K index:
__device__ __forceinline__ int kmapA(int e, int lane) {
  int j = e >> 1, pos = e & 1;
  return ((j & 3) << 1) + pos + ((lane >> 4) << 3) + ((j >> 2) << 4);
}
// B-operand: lane holds col n = lane&15; element e -> K = (e&7) + 8*(lane>=16) + 16*(e>=8)
__device__ __forceinline__ int kmapB(int e, int lane) {
  return (e & 7) + ((lane >> 4) << 3) + ((e >> 3) << 4);
}
// slope 0.01 > 0  =>  leaky-relu == max(x, 0.01x): one mul + one max, no cmp/cndmask
__device__ __forceinline__ float lrelu(float x) { return fmaxf(x, 0.01f * x); }

__device__ __forceinline__ v16h lds_a_frag(const _Float16* tile, int ld, int kbase, int lane) {
  const _Float16* row = tile + (lane & 15) * ld + kbase;
  v16h a;
#pragma unroll
  for (int e = 0; e < 16; ++e) a[e] = row[kmapA(e, lane)];
  return a;
}

// B fragments are pre-packed: frag buffer[frag][lane][16 halves] -> two b128 loads
__device__ __forceinline__ v16h load_b_frag(const _Float16* base, int frag, int lane) {
  const uint4* p = (const uint4*)(base + (((size_t)frag << 5) + lane) * 16);
  union { uint4 q[2]; v16h h; } cv;
  cv.q[0] = p[0];
  cv.q[1] = p[1];
  return cv.h;
}

// D fragment (f32) -> f16 LDS tile, optional leaky-ReLU. m = r + 8*(lane>=16), n = ncol + lane&15
__device__ __forceinline__ void store_frag_lds(v8f acc, _Float16* tile, int ld, int ncol,
                                               int lane, bool act) {
  int n  = ncol + (lane & 15);
  int mb = (lane >> 4) << 3;
#pragma unroll
  for (int r = 0; r < 8; ++r) {
    float v = acc[r];
    if (act) v = lrelu(v);
    tile[(mb + r) * ld + n] = (_Float16)v;
  }
}

// copy a 16x128 f16 tile (contiguous in both LDS and global) with coalesced b128s
__device__ __forceinline__ void copy_tile_out(const _Float16* src, _Float16* dst, int lane) {
  const uint4* s = (const uint4*)src;
  uint4* d = (uint4*)dst;
#pragma unroll
  for (int i = 0; i < 8; ++i) d[lane + i * 32] = s[lane + i * 32];
}

#define WMMA_F16(acc, av, bv) \
  __builtin_amdgcn_wmma_f32_16x16x32_f16(false, (av), false, (bv), (short)0, (acc), false, false)

// ---------------- kernel 0: convert weights to fragment-ready f16 ----------------
__global__ __launch_bounds__(256) void k_prep(
    const float* __restrict__ enc_W, const float* __restrict__ s_W,
    const float* __restrict__ key_W, const float* __restrict__ sel_W,
    const float* __restrict__ val_W, const float* __restrict__ crit_W1,
    const float* __restrict__ crit_W2, _Float16* __restrict__ ws) {
  long idx = (long)blockIdx.x * 256 + threadIdx.x;
  long li = idx;
  if (li < (long)N_ENCW) {  // enc_W: [A][IDIM][H], K padded 80->96 (3 K-tiles), 8 N-tiles
    int f = (int)(li >> 9), r = (int)(li & 511), lane = r >> 4, e = r & 15;
    int a = f / 24, kt = (f >> 3) % 3, nt = f & 7;
    int k = kt * 32 + kmapB(e, lane), n = nt * 16 + (lane & 15);
    float v = (k < IDIM) ? enc_W[((size_t)a * IDIM + k) * H + n] : 0.f;
    ws[OFF_ENCW + li] = (_Float16)v;
    return;
  }
  li -= (long)N_ENCW;
  if (li < (long)N_SW) {    // s_W: [A][OBS][H], 2 K-tiles
    int f = (int)(li >> 9), r = (int)(li & 511), lane = r >> 4, e = r & 15;
    int a = f >> 4, kt = (f >> 3) & 1, nt = f & 7;
    int k = kt * 32 + kmapB(e, lane), n = nt * 16 + (lane & 15);
    ws[OFF_SW + li] = (_Float16)s_W[((size_t)a * OBS + k) * H + n];
    return;
  }
  li -= (long)N_SW;
  if (li < (long)(3 * N_HW)) {  // key/sel/val W: [HEADS][H][DD] -> K=H, N=ed (head-major)
    int which = (int)(li / (long)N_HW);
    long lj = li % (long)N_HW;
    int f = (int)(lj >> 9), r = (int)(lj & 511), lane = r >> 4, e = r & 15;
    int kt = f >> 3, nt = f & 7;
    int k = kt * 32 + kmapB(e, lane);
    int ed = nt * 16 + (lane & 15);
    int hh = ed >> 5, d = ed & 31;
    const float* src = (which == 0) ? key_W : ((which == 1) ? sel_W : val_W);
    size_t off = (which == 0) ? OFF_KEYW : ((which == 1) ? OFF_SELW : OFF_VALW);
    float v = src[((size_t)hh * H + k) * DD + d];
    if (which == 1) v *= ATTN_SCALE;      // fold 1/sqrt(D) into selector weights
    ws[off + lj] = (_Float16)v;
    return;
  }
  li -= (long)(3 * N_HW);
  if (li < (long)N_CW1) {   // crit_W1: [A][2H][H], 8 K-tiles
    int f = (int)(li >> 9), r = (int)(li & 511), lane = r >> 4, e = r & 15;
    int a = f >> 6, kt = (f >> 3) & 7, nt = f & 7;
    int k = kt * 32 + kmapB(e, lane), n = nt * 16 + (lane & 15);
    ws[OFF_CW1 + li] = (_Float16)crit_W1[((size_t)a * (2 * H) + k) * H + n];
    return;
  }
  li -= (long)N_CW1;
  if (li < (long)N_CW2) {   // crit_W2: [A][H][ACT], 4 K-tiles, N=16
    int f = (int)(li >> 9), r = (int)(li & 511), lane = r >> 4, e = r & 15;
    int a = f >> 2, kt = f & 3;
    int k = kt * 32 + kmapB(e, lane), n = lane & 15;
    ws[OFF_CW2 + li] = (_Float16)crit_W2[((size_t)a * H + k) * ACT + n];
    return;
  }
}

// ---------------- kernel 1: encoders + per-head projections ----------------
__global__ __launch_bounds__(128) void k_encode(
    const float* __restrict__ obs, const float* __restrict__ u,
    const float* __restrict__ enc_b, const float* __restrict__ s_b,
    const float* __restrict__ val_b, _Float16* __restrict__ ws) {
  __shared__ _Float16 inT[4][16][96];
  __shared__ _Float16 saT[4][16][H];
  __shared__ _Float16 seT[4][16][H];
  __shared__ _Float16 outT[4][16][H];
  const int wave = threadIdx.x >> 5, lane = threadIdx.x & 31;
  const int a = blockIdx.x >> 9;                     // grid = A * 512
  const int token0 = ((blockIdx.x & 511) * 4 + wave) * 16;
  const size_t tokBase = (size_t)a * NTOK + token0;
  const float* obsA = obs + tokBase * OBS;
  const float* uA   = u   + tokBase * ACT;

  // stage input tile: [16][96] f16 = concat(obs, u, zero-pad)
  for (int r = 0; r < 16; ++r)
    for (int c = lane; c < 96; c += 32) {
      float v = (c < OBS) ? obsA[r * OBS + c]
                          : ((c < IDIM) ? uA[r * ACT + (c - OBS)] : 0.f);
      inT[wave][r][c] = (_Float16)v;
    }
  __syncthreads();

  const _Float16* inP = &inT[wave][0][0];
  _Float16* saP = &saT[wave][0][0];
  _Float16* seP = &seT[wave][0][0];
  _Float16* outP = &outT[wave][0][0];

  // sa = lrelu(inp @ enc_W + enc_b)   (K = 96 padded, 3 WMMA per N-chunk)
  {
    const _Float16* wf = ws + OFF_ENCW + (size_t)a * (3 * 8 * 512);
    for (int nt = 0; nt < 8; ++nt) {
      float b0 = enc_b[a * H + nt * 16 + (lane & 15)];
      v8f acc;
#pragma unroll
      for (int r = 0; r < 8; ++r) acc[r] = b0;
#pragma unroll
      for (int kt = 0; kt < 3; ++kt) {
        v16h av = lds_a_frag(inP, 96, kt * 32, lane);
        v16h bv = load_b_frag(wf, kt * 8 + nt, lane);
        acc = WMMA_F16(acc, av, bv);
      }
      store_frag_lds(acc, saP, H, nt * 16, lane, true);
    }
  }
  // se = lrelu(obs @ s_W + s_b)   (K = 64)
  {
    const _Float16* wf = ws + OFF_SW + (size_t)a * (2 * 8 * 512);
    for (int nt = 0; nt < 8; ++nt) {
      float b0 = s_b[a * H + nt * 16 + (lane & 15)];
      v8f acc;
#pragma unroll
      for (int r = 0; r < 8; ++r) acc[r] = b0;
#pragma unroll
      for (int kt = 0; kt < 2; ++kt) {
        v16h av = lds_a_frag(inP, 96, kt * 32, lane);
        v16h bv = load_b_frag(wf, kt * 8 + nt, lane);
        acc = WMMA_F16(acc, av, bv);
      }
      store_frag_lds(acc, seP, H, nt * 16, lane, true);
    }
  }
  copy_tile_out(seP, ws + OFF_SE + tokBase * H, lane);

  // keys = sa @ key_W (K = 128, head-concat columns)
  {
    const _Float16* wf = ws + OFF_KEYW;
    for (int nt = 0; nt < 8; ++nt) {
      v8f acc;
#pragma unroll
      for (int r = 0; r < 8; ++r) acc[r] = 0.f;
#pragma unroll
      for (int kt = 0; kt < 4; ++kt) {
        v16h av = lds_a_frag(saP, H, kt * 32, lane);
        v16h bv = load_b_frag(wf, kt * 8 + nt, lane);
        acc = WMMA_F16(acc, av, bv);
      }
      store_frag_lds(acc, outP, H, nt * 16, lane, false);
    }
    copy_tile_out(outP, ws + OFF_KEYS + tokBase * H, lane);
  }
  // sel = se @ (sel_W / sqrt(D))
  {
    const _Float16* wf = ws + OFF_SELW;
    for (int nt = 0; nt < 8; ++nt) {
      v8f acc;
#pragma unroll
      for (int r = 0; r < 8; ++r) acc[r] = 0.f;
#pragma unroll
      for (int kt = 0; kt < 4; ++kt) {
        v16h av = lds_a_frag(seP, H, kt * 32, lane);
        v16h bv = load_b_frag(wf, kt * 8 + nt, lane);
        acc = WMMA_F16(acc, av, bv);
      }
      store_frag_lds(acc, outP, H, nt * 16, lane, false);
    }
    copy_tile_out(outP, ws + OFF_SEL + tokBase * H, lane);
  }
  // vals = lrelu(sa @ val_W + val_b)
  {
    const _Float16* wf = ws + OFF_VALW;
    for (int nt = 0; nt < 8; ++nt) {
      float b0 = val_b[nt * 16 + (lane & 15)];   // [HEADS*DD] contiguous, head-major
      v8f acc;
#pragma unroll
      for (int r = 0; r < 8; ++r) acc[r] = b0;
#pragma unroll
      for (int kt = 0; kt < 4; ++kt) {
        v16h av = lds_a_frag(saP, H, kt * 32, lane);
        v16h bv = load_b_frag(wf, kt * 8 + nt, lane);
        acc = WMMA_F16(acc, av, bv);
      }
      store_frag_lds(acc, outP, H, nt * 16, lane, true);
    }
    copy_tile_out(outP, ws + OFF_VALS + tokBase * H, lane);
  }
}

// ---------------- kernel 2: self-exclusion attention over agents ----------------
__global__ __launch_bounds__(128) void k_attn(_Float16* __restrict__ ws) {
  __shared__ _Float16 selS[4][A_N][H];
  __shared__ _Float16 keyS[4][A_N][H];
  __shared__ _Float16 valS[4][A_N][H];
  __shared__ _Float16 othS[4][A_N][H];
  const int wave = threadIdx.x >> 5, lane = threadIdx.x & 31;
  const int n = blockIdx.x * 4 + wave;               // token id (b*T + t)
  const _Float16* selG = ws + OFF_SEL;
  const _Float16* keyG = ws + OFF_KEYS;
  const _Float16* valG = ws + OFF_VALS;
  _Float16* othG = ws + OFF_OTHER;

  for (int idx = lane; idx < A_N * 16; idx += 32) {  // 16 uint4 per agent row
    int ag = idx >> 4, q = idx & 15;
    size_t off = ((size_t)ag * NTOK + n) * H;
    ((uint4*)&selS[wave][ag][0])[q] = ((const uint4*)(selG + off))[q];
    ((uint4*)&keyS[wave][ag][0])[q] = ((const uint4*)(keyG + off))[q];
    ((uint4*)&valS[wave][ag][0])[q] = ((const uint4*)(valG + off))[q];
  }
  __syncthreads();

  const int i = lane >> 2, e = lane & 3;             // lane = (agent, head)
  float sel_r[DD];
#pragma unroll
  for (int d = 0; d < DD; ++d) sel_r[d] = (float)selS[wave][i][e * DD + d];

  float lg[A_N];
#pragma unroll
  for (int j = 0; j < A_N; ++j) {
    float s = 0.f;
#pragma unroll
    for (int d = 0; d < DD; ++d) s += sel_r[d] * (float)keyS[wave][j][e * DD + d];
    lg[j] = (j == i) ? -1e9f : s;                    // scale pre-folded into sel_W
  }
  float mx = lg[0];
#pragma unroll
  for (int j = 1; j < A_N; ++j) mx = fmaxf(mx, lg[j]);
  float sum = 0.f;
#pragma unroll
  for (int j = 0; j < A_N; ++j) { lg[j] = __expf(lg[j] - mx); sum += lg[j]; }
  float inv = 1.f / sum;
#pragma unroll
  for (int j = 0; j < A_N; ++j) lg[j] *= inv;

#pragma unroll
  for (int d = 0; d < DD; ++d) {
    float o = 0.f;
#pragma unroll
    for (int j = 0; j < A_N; ++j) o += lg[j] * (float)valS[wave][j][e * DD + d];
    othS[wave][i][e * DD + d] = (_Float16)o;
  }
  __syncthreads();
  for (int idx = lane; idx < A_N * 16; idx += 32) {
    int ag = idx >> 4, q = idx & 15;
    ((uint4*)(othG + ((size_t)ag * NTOK + n) * H))[q] = ((const uint4*)&othS[wave][ag][0])[q];
  }
}

// ---------------- kernel 3: critic MLP + action-Q gather ----------------
__global__ __launch_bounds__(128) void k_critic(
    const float* __restrict__ u, const float* __restrict__ crit_b1,
    const float* __restrict__ crit_b2, const _Float16* __restrict__ ws,
    float* __restrict__ out) {
  __shared__ _Float16 cT[4][16][2 * H];
  __shared__ _Float16 h1T[4][16][H];
  __shared__ float    qT[4][16][ACT];
  const int wave = threadIdx.x >> 5, lane = threadIdx.x & 31;
  const int a = blockIdx.x >> 9;
  const int token0 = ((blockIdx.x & 511) * 4 + wave) * 16;
  const size_t tokBase = (size_t)a * NTOK + token0;

  // stage critic_in = [se | other] : [16][256] f16
  const uint4* seS = (const uint4*)(ws + OFF_SE + tokBase * H);
  const uint4* otS = (const uint4*)(ws + OFF_OTHER + tokBase * H);
  for (int idx = lane; idx < 256; idx += 32) {       // 256 uint4 per 16x128 source
    int r = idx >> 4, c = idx & 15;
    ((uint4*)&cT[wave][r][0])[c]      = seS[idx];
    ((uint4*)&cT[wave][r][0])[16 + c] = otS[idx];
  }
  __syncthreads();

  const _Float16* cP = &cT[wave][0][0];
  _Float16* hP = &h1T[wave][0][0];

  // h1 = lrelu(critic_in @ crit_W1 + b1)   (K = 256)
  {
    const _Float16* wf = ws + OFF_CW1 + (size_t)a * (8 * 8 * 512);
    for (int nt = 0; nt < 8; ++nt) {
      float b0 = crit_b1[a * H + nt * 16 + (lane & 15)];
      v8f acc;
#pragma unroll
      for (int r = 0; r < 8; ++r) acc[r] = b0;
#pragma unroll
      for (int kt = 0; kt < 8; ++kt) {
        v16h av = lds_a_frag(cP, 2 * H, kt * 32, lane);
        v16h bv = load_b_frag(wf, kt * 8 + nt, lane);
        acc = WMMA_F16(acc, av, bv);
      }
      store_frag_lds(acc, hP, H, nt * 16, lane, true);
    }
  }
  // all_q = h1 @ crit_W2 + b2   (K = 128, N = 16)
  {
    const _Float16* wf = ws + OFF_CW2 + (size_t)a * (4 * 512);
    float b0 = crit_b2[a * ACT + (lane & 15)];
    v8f acc;
#pragma unroll
    for (int r = 0; r < 8; ++r) acc[r] = b0;
#pragma unroll
    for (int kt = 0; kt < 4; ++kt) {
      v16h av = lds_a_frag(hP, H, kt * 32, lane);
      v16h bv = load_b_frag(wf, kt, lane);
      acc = WMMA_F16(acc, av, bv);
    }
    int nn = lane & 15, mb = (lane >> 4) << 3;
#pragma unroll
    for (int r = 0; r < 8; ++r) qT[wave][mb + r][nn] = acc[r];
  }
  __syncthreads();

  // q = all_q[argmax(u)]  (first-max, like jnp.argmax)
  if (lane < 16) {
    int tok = token0 + lane;
    const float* uu = u + ((size_t)a * NTOK + tok) * ACT;
    int best = 0;
    float bv = uu[0];
#pragma unroll
    for (int c = 1; c < ACT; ++c) {
      float v = uu[c];
      if (v > bv) { bv = v; best = c; }
    }
    out[(size_t)a * NTOK + tok] = qT[wave][lane][best];
  }
}

// ---------------- host launcher ----------------
extern "C" void kernel_launch(void* const* d_in, const int* in_sizes, int n_in,
                              void* d_out, int out_size, void* d_ws, size_t ws_size,
                              hipStream_t stream) {
  (void)in_sizes; (void)n_in; (void)out_size; (void)ws_size;
  const float* obs     = (const float*)d_in[0];
  const float* u       = (const float*)d_in[1];
  const float* enc_W   = (const float*)d_in[2];
  const float* enc_b   = (const float*)d_in[3];
  const float* s_W     = (const float*)d_in[4];
  const float* s_b     = (const float*)d_in[5];
  const float* key_W   = (const float*)d_in[6];
  const float* sel_W   = (const float*)d_in[7];
  const float* val_W   = (const float*)d_in[8];
  const float* val_b   = (const float*)d_in[9];
  const float* crit_W1 = (const float*)d_in[10];
  const float* crit_b1 = (const float*)d_in[11];
  const float* crit_W2 = (const float*)d_in[12];
  const float* crit_b2 = (const float*)d_in[13];
  _Float16* ws = (_Float16*)d_ws;
  float* out = (float*)d_out;

  const long prep_elems = (long)(N_ENCW + N_SW + 3 * N_HW + N_CW1 + N_CW2); // 491520
  const int prep_blocks = (int)((prep_elems + 255) / 256);                  // 1920

  k_prep<<<prep_blocks, 256, 0, stream>>>(enc_W, s_W, key_W, sel_W, val_W,
                                          crit_W1, crit_W2, ws);
  k_encode<<<A_N * (NTOK / 64), 128, 0, stream>>>(obs, u, enc_b, s_b, val_b, ws); // 4096 blocks
  k_attn<<<NTOK / 4, 128, 0, stream>>>(ws);                                       // 8192 blocks
  k_critic<<<A_N * (NTOK / 64), 128, 0, stream>>>(u, crit_b1, crit_b2, ws, out);  // 4096 blocks
}
